// SRUCell_64776696758612
// MI455X (gfx1250) — matrix operand order repaired and checked by reading, not compiled, not run on hardware
//
#include <hip/hip_runtime.h>

typedef __attribute__((ext_vector_type(16))) __bf16 v16bf;
typedef __attribute__((ext_vector_type(8)))  __bf16 v8bf;
typedef __attribute__((ext_vector_type(4)))  __bf16 v4bf;
typedef __attribute__((ext_vector_type(8)))  float  v8f;
typedef __attribute__((ext_vector_type(4)))  float  v4f;

#define L_SEQ 2048
#define BATCH 16
#define DH    1024
#define MROWS (L_SEQ * BATCH)   // 32768
#define KDIM  DH                // 1024
#define NDIM  (3 * DH)          // 3072

#define LOG2E  1.4426950408889634f
#define LOG2E2 2.8853900817779268f

// ---------------------------------------------------------------------------
// Kernel 1: convert input X (M x K fp32, row-major) -> bf16, vectorized x4
// ---------------------------------------------------------------------------
__global__ __launch_bounds__(256) void cvt_x_kernel(const float* __restrict__ x,
                                                    __bf16* __restrict__ xb,
                                                    int n4) {
  int i = blockIdx.x * blockDim.x + threadIdx.x;
  if (i >= n4) return;
  v4f v = ((const v4f*)x)[i];
  v4bf o;
  o.x = (__bf16)v.x;
  o.y = (__bf16)v.y;
  o.z = (__bf16)v.z;
  o.w = (__bf16)v.w;
  ((v4bf*)xb)[i] = o;
}

// ---------------------------------------------------------------------------
// Kernel 2: convert + transpose W: W[k][n] fp32 -> Wt[n][k] bf16
// ---------------------------------------------------------------------------
__global__ __launch_bounds__(256) void cvt_w_kernel(const float* __restrict__ w,
                                                    __bf16* __restrict__ wt) {
  int idx = blockIdx.x * blockDim.x + threadIdx.x;  // over K*N
  if (idx >= KDIM * NDIM) return;
  int k = idx / NDIM;
  int n = idx - k * NDIM;
  wt[(size_t)n * KDIM + k] = (__bf16)w[idx];
}

// ---------------------------------------------------------------------------
// Fragment helpers
// ---------------------------------------------------------------------------
__device__ __forceinline__ v16bf ld_frag(const __bf16* p_lo,
                                         const __bf16* p_hi) {
  v8bf lo = *(const v8bf*)p_lo;
  v8bf hi = *(const v8bf*)p_hi;
  return __builtin_shufflevector(lo, hi, 0, 1, 2, 3, 4, 5, 6, 7, 8, 9, 10, 11,
                                 12, 13, 14, 15);
}

struct Frags {
  v16bf a0, a1, b0, b1, b2, b3;
};

__device__ __forceinline__ Frags load_frags(const __bf16* aRow0,
                                            const __bf16* aRow1,
                                            const __bf16* bRow0,
                                            const __bf16* bRow1,
                                            const __bf16* bRow2,
                                            const __bf16* bRow3, int k0) {
  Frags f;
  f.a0 = ld_frag(aRow0 + k0, aRow0 + k0 + 8);
  f.a1 = ld_frag(aRow1 + k0, aRow1 + k0 + 8);
  f.b0 = ld_frag(bRow0 + k0, bRow0 + k0 + 8);
  f.b1 = ld_frag(bRow1 + k0, bRow1 + k0 + 8);
  f.b2 = ld_frag(bRow2 + k0, bRow2 + k0 + 8);
  f.b3 = ld_frag(bRow3 + k0, bRow3 + k0 + 8);
  return f;
}

__device__ __forceinline__ void wmma8(v8f acc[2][4], const Frags& f) {
  acc[0][0] = __builtin_amdgcn_wmma_f32_16x16x32_bf16(
      false, f.a0, false, f.b0, (short)0, acc[0][0], false, false);
  acc[0][1] = __builtin_amdgcn_wmma_f32_16x16x32_bf16(
      false, f.a0, false, f.b1, (short)0, acc[0][1], false, false);
  acc[0][2] = __builtin_amdgcn_wmma_f32_16x16x32_bf16(
      false, f.a0, false, f.b2, (short)0, acc[0][2], false, false);
  acc[0][3] = __builtin_amdgcn_wmma_f32_16x16x32_bf16(
      false, f.a0, false, f.b3, (short)0, acc[0][3], false, false);
  acc[1][0] = __builtin_amdgcn_wmma_f32_16x16x32_bf16(
      false, f.a1, false, f.b0, (short)0, acc[1][0], false, false);
  acc[1][1] = __builtin_amdgcn_wmma_f32_16x16x32_bf16(
      false, f.a1, false, f.b1, (short)0, acc[1][1], false, false);
  acc[1][2] = __builtin_amdgcn_wmma_f32_16x16x32_bf16(
      false, f.a1, false, f.b2, (short)0, acc[1][2], false, false);
  acc[1][3] = __builtin_amdgcn_wmma_f32_16x16x32_bf16(
      false, f.a1, false, f.b3, (short)0, acc[1][3], false, false);
}

__device__ __forceinline__ float fast_sigmoid_like(float u, bool isTanh) {
  // sigmoid(u) = rcp(1 + exp2(-u*log2e)) ; tanh(u) = 2*sigmoid(2u) - 1
  float c = isTanh ? -LOG2E2 : -LOG2E;
  float e = __builtin_amdgcn_exp2f(u * c);
  float s = __builtin_amdgcn_rcpf(1.0f + e);
  return isTanh ? __builtin_fmaf(2.0f, s, -1.0f) : s;
}

// ---------------------------------------------------------------------------
// Kernel 3: GEMM U = X @ W via v_wmma_f32_16x16x32_bf16.
// Wave computes 32x64 tile: 2 A frags x 4 B frags = 8 WMMAs per k-step of 32.
// Ping-pong pipelining (k-loop unrolled x2, two independent fragment sets ->
// no inter-buffer register copies, loads overlap WMMA execution).
//
// Fragment layouts used here (wave32; same contiguous K order in A and B):
//  A (16x32 bf16): lane L holds row M=L%16; lanes 0-15 carry K = 0..15;
//                  lanes 16-31 carry K = 16..31.
//  B (32x16 bf16): lane L holds column N=L%16; lanes 0-15 carry K = 0..15;
//                  lanes 16-31 carry K = 16..31.  (Wt row-major in K)
//  C/D (16x16 f32): VGPR e holds M = e + (lane>=16 ? 8 : 0), N = lane%16.
// ---------------------------------------------------------------------------
__global__ __launch_bounds__(128) void sru_gemm_kernel(
    const __bf16* __restrict__ Xb, const __bf16* __restrict__ Wt,
    const float* __restrict__ bias, float* __restrict__ Uact) {
  const int wave = threadIdx.x >> 5;
  const int lane = threadIdx.x & 31;
  const int gw = blockIdx.x * 4 + wave;

  const int NT = NDIM / 64;  // 48 N-tiles of width 64
  const int mTile = gw / NT;
  const int nTile = gw - mTile * NT;
  const int rowBase = mTile * 32;
  const int colBase = nTile * 64;

  const int lh = lane & 15;
  const int hi = lane >> 4;  // 0 or 1 (half-wave select)

  const __bf16* aRow0 = Xb + (size_t)(rowBase + lh) * KDIM + hi * 16;
  const __bf16* aRow1 = aRow0 + (size_t)16 * KDIM;
  const __bf16* bRow0 = Wt + (size_t)(colBase + lh) * KDIM + hi * 16;
  const __bf16* bRow1 = bRow0 + (size_t)16 * KDIM;
  const __bf16* bRow2 = bRow0 + (size_t)32 * KDIM;
  const __bf16* bRow3 = bRow0 + (size_t)48 * KDIM;

  v8f acc[2][4] = {{v8f{}, v8f{}, v8f{}, v8f{}}, {v8f{}, v8f{}, v8f{}, v8f{}}};

  // prologue: stage 0
  Frags f0 = load_frags(aRow0, aRow1, bRow0, bRow1, bRow2, bRow3, 0);
  Frags f1;

#pragma unroll 1
  for (int k0 = 0; k0 < KDIM - 64; k0 += 64) {
    f1 = load_frags(aRow0, aRow1, bRow0, bRow1, bRow2, bRow3, k0 + 32);
    wmma8(acc, f0);
    f0 = load_frags(aRow0, aRow1, bRow0, bRow1, bRow2, bRow3, k0 + 64);
    wmma8(acc, f1);
  }
  // tail: f0 holds k = KDIM-64
  f1 = load_frags(aRow0, aRow1, bRow0, bRow1, bRow2, bRow3, KDIM - 32);
  wmma8(acc, f0);
  wmma8(acc, f1);

  // ---- fused bias + activation epilogue (branchless fast tanh/sigmoid) --
#pragma unroll
  for (int mi = 0; mi < 2; ++mi) {
#pragma unroll
    for (int j = 0; j < 4; ++j) {
      const int n = colBase + j * 16 + lh;
      const float bv = bias[n];
      const bool isTanh = (n < DH);
#pragma unroll
      for (int e = 0; e < 8; ++e) {
        const int mrow = rowBase + mi * 16 + e + hi * 8;
        const float u = acc[mi][j][e] + bv;
        Uact[(size_t)mrow * NDIM + n] = fast_sigmoid_like(u, isTanh);
      }
    }
  }
}

// ---------------------------------------------------------------------------
// Kernel 4: sequential scan along L, one thread per (b, d) channel.
// c_t = f_t*c_{t-1} + (1-f_t)*xt ;  h_t = r_t*c_t + (1-r_t)*input_t
// ---------------------------------------------------------------------------
__global__ __launch_bounds__(256) void sru_scan_kernel(
    const float* __restrict__ Uact, const float* __restrict__ X,
    const float* __restrict__ c0, float* __restrict__ hOut,
    float* __restrict__ cOut) {
  const int ch = blockIdx.x * blockDim.x + threadIdx.x;  // 0 .. B*D-1
  const int d = ch & (DH - 1);
  const int b = ch >> 10;

  float c = c0[ch];  // c0 is (B, D) row-major: b*DH + d == ch
  for (int t = 0; t < L_SEQ; ++t) {
    const size_t m = (size_t)t * BATCH + b;
    const size_t ub = m * NDIM + d;
    const float xt = Uact[ub];
    const float f = Uact[ub + DH];
    const float r = Uact[ub + 2 * DH];
    const float inp = X[m * DH + d];
    c = f * c + (1.0f - f) * xt;
    const float h = r * c + (1.0f - r) * inp;
    const size_t o = m * DH + d;
    hOut[o] = h;
    cOut[o] = c;
  }
}

// ---------------------------------------------------------------------------
extern "C" void kernel_launch(void* const* d_in, const int* in_sizes, int n_in,
                              void* d_out, int out_size, void* d_ws,
                              size_t ws_size, hipStream_t stream) {
  const float* x = (const float*)d_in[0];     // (L, B, D)
  const float* c0 = (const float*)d_in[1];    // (B, D)
  const float* w = (const float*)d_in[2];     // (D, 3D)
  const float* bias = (const float*)d_in[3];  // (3D,)

  float* hOut = (float*)d_out;
  float* cOut = hOut + (size_t)L_SEQ * BATCH * DH;

  // Workspace layout:
  //   Xb   : MROWS*KDIM bf16   (64 MB)
  //   Wt   : NDIM*KDIM  bf16   ( 6 MB)
  //   Uact : MROWS*NDIM fp32   (384 MB)
  char* ws = (char*)d_ws;
  __bf16* Xb = (__bf16*)ws;
  __bf16* Wt = (__bf16*)(ws + (size_t)MROWS * KDIM * sizeof(__bf16));
  float* Uact = (float*)(ws + (size_t)MROWS * KDIM * sizeof(__bf16) +
                         (size_t)NDIM * KDIM * sizeof(__bf16));

  const int n4 = MROWS * KDIM / 4;
  cvt_x_kernel<<<(n4 + 255) / 256, 256, 0, stream>>>(x, Xb, n4);

  const int nw = KDIM * NDIM;
  cvt_w_kernel<<<(nw + 255) / 256, 256, 0, stream>>>(w, Wt);

  const int totalWaves = (MROWS / 32) * (NDIM / 64);  // 1024 * 48
  sru_gemm_kernel<<<totalWaves / 4, 128, 0, stream>>>(Xb, Wt, bias, Uact);

  sru_scan_kernel<<<(BATCH * DH) / 256, 256, 0, stream>>>(Uact, x, c0, hOut,
                                                          cOut);
}
